// SelfAttention_51788715655173
// MI455X (gfx1250) — matrix operand order repaired and checked by reading, not compiled, not run on hardware
//
#include <hip/hip_runtime.h>

typedef _Float16 v16h __attribute__((ext_vector_type(16)));
typedef float    v8f  __attribute__((ext_vector_type(8)));
typedef int      v4i  __attribute__((vector_size(4 * sizeof(int))));

#define D_MODEL 1024
#define D_K     128
#define BATCH   8
#define SEQ     2048
#define MTOT    (BATCH * SEQ)

union AFrag { v16h v; _Float16 h[16]; uint4 q[2]; };
union CFrag { v8f  v; float    f[8];  };

// A-matrix (16x32 f16) per-lane K offset for element e (ISA 7.12.2):
//   lanes 0-15 : e<8 -> K=e,   e>=8 -> K=16+(e-8)
//   lanes 16-31: e<8 -> K=8+e, e>=8 -> K=24+(e-8)
// => element block [0..7] starts at K = hi?8:0, block [8..15] at K = hi?24:16.

// ---- gfx1250 async global->LDS copy (ASYNCcnt path), with sync fallback ----
#if __has_builtin(__builtin_amdgcn_global_load_async_to_lds_b128)
#define HAVE_ASYNC_COPY 1
typedef __attribute__((address_space(1))) v4i* gv4i_p;   // AS1 ("__device__")
typedef __attribute__((address_space(3))) v4i* lv4i_p;   // AS3 ("__shared__")
#else
#define HAVE_ASYNC_COPY 0
#endif

__device__ __forceinline__ void cp16_g2l(_Float16* l, const _Float16* g) {
#if HAVE_ASYNC_COPY
  __builtin_amdgcn_global_load_async_to_lds_b128(
      (gv4i_p)(_Float16*)g, (lv4i_p)l, 0, 0);
#else
  *(uint4*)l = *(const uint4*)g;
#endif
}

__device__ __forceinline__ void wait_async_all() {
#if HAVE_ASYNC_COPY
#if __has_builtin(__builtin_amdgcn_s_wait_asynccnt)
  __builtin_amdgcn_s_wait_asynccnt(0);
#else
  asm volatile("s_wait_asynccnt 0x0" ::: "memory");
#endif
#endif
}

// ---------------------------------------------------------------------------
// Kernel 1: fused QKV projection.  X[16384,1024]f32 x W[128,1024]f32^T ->
// Q,K : [16384,128] f16 (row-major tokens)
// V   : transposed per batch, V^T[b][n=128][t=2048] f16  (so the attention
//       kernel can both async-copy V tiles AND read B-fragments as b128s)
// grid=(M/16, 3), block=256 (8 waves, one 16-col N-tile each)
// ---------------------------------------------------------------------------
__global__ __launch_bounds__(256) void qkv_project_kernel(
    const float* __restrict__ X,
    const float* __restrict__ Wq, const float* __restrict__ Wk,
    const float* __restrict__ Wv,
    _Float16* __restrict__ Qh, _Float16* __restrict__ Kh,
    _Float16* __restrict__ Vt)
{
  __shared__ _Float16 xs[16 * 1032];            // 16x1024 tile, padded rows
  const int mBase = blockIdx.x * 16;
  const int mat   = blockIdx.y;
  const float* W = (mat == 0) ? Wq : (mat == 1) ? Wk : Wv;

  for (int idx = threadIdx.x; idx < 16 * D_MODEL; idx += 256) {
    const int m = idx >> 10, c = idx & 1023;
    xs[m * 1032 + c] = (_Float16)X[(size_t)(mBase + m) * D_MODEL + c];
  }
  __syncthreads();

  const int lane  = threadIdx.x & 31;
  const int wave  = threadIdx.x >> 5;
  const int ln    = lane & 15;
  const int hi    = lane >> 4;
  const int nBase = wave * 16;

  CFrag c;
#pragma unroll
  for (int r = 0; r < 8; ++r) c.f[r] = 0.0f;

  for (int k0 = 0; k0 < D_MODEL; k0 += 32) {
    AFrag a, b;
    const _Float16* xrow = xs + ln * 1032 + k0;       // 16B-aligned segments
    a.q[0] = *(const uint4*)(xrow + (hi ? 8 : 0));
    a.q[1] = *(const uint4*)(xrow + (hi ? 24 : 16));
    const float* wrow = W + (size_t)(nBase + ln) * D_MODEL + k0 + (hi ? 8 : 0);
#pragma unroll
    for (int e = 0; e < 16; ++e)
      b.h[e] = (_Float16)wrow[e < 8 ? e : e + 8];
    c.v = __builtin_amdgcn_wmma_f32_16x16x32_f16(false, a.v, false, b.v,
                                                 (short)0, c.v, false, false);
  }

  if (mat == 2) {                               // V: store transposed
#pragma unroll
    for (int r = 0; r < 8; ++r) {
      const int row = mBase + r + (hi ? 8 : 0); // global token index
      const int bb  = row >> 11;                // batch
      const int t   = row & (SEQ - 1);          // token within batch
      Vt[(size_t)bb * D_K * SEQ + (size_t)(nBase + ln) * SEQ + t] =
          (_Float16)c.f[r];
    }
  } else {
    _Float16* Out = (mat == 0) ? Qh : Kh;
#pragma unroll
    for (int r = 0; r < 8; ++r) {
      const int row = mBase + r + (hi ? 8 : 0);
      Out[(size_t)row * D_K + nBase + ln] = (_Float16)c.f[r];
    }
  }
}

// ---------------------------------------------------------------------------
// Kernel 2: causal flash attention.
// Block = 128 threads = 4 waves = 4 consecutive 16-row q tiles (64 rows).
// K (row-major) and V^T tiles double-buffered in LDS via async global->LDS
// b128 copies; all fragment builds are b128 LDS/global reads.  Per key block
// per wave: 8 WMMAs (scores) + 8 WMMAs (PV).
// ---------------------------------------------------------------------------
__global__ __launch_bounds__(128) void flash_attn_kernel(
    const _Float16* __restrict__ Qh, const _Float16* __restrict__ Kh,
    const _Float16* __restrict__ Vt, float* __restrict__ Out)
{
  __shared__ _Float16 ks[2][32 * 128];          // 2 x 8KB K tiles [key][feat]
  __shared__ _Float16 vsT[2][128 * 40];         // 2 x 10KB V^T tiles [n][40]
  __shared__ _Float16 ps[4][16 * 40];           // per-wave P transpose bounce

  const int tid  = threadIdx.x;
  const int wave = tid >> 5;
  const int lane = tid & 31;
  const int ln   = lane & 15;
  const int hi   = lane >> 4;
  const int b    = blockIdx.y;
  const int qt   = blockIdx.x * 4 + wave;       // this wave's q tile
  const int q0   = qt * 16;
  const int qtmax = blockIdx.x * 4 + 3;
  const int nb   = (qtmax + 2) >> 1;            // block-uniform causal bound

  const _Float16* Qb  = Qh + (size_t)b * SEQ * D_K;
  const _Float16* Kb  = Kh + (size_t)b * SEQ * D_K;
  const _Float16* VbT = Vt + (size_t)b * D_K * SEQ;

  // Cooperative K + V^T tile stage: 512+512 x 16B chunks, 4+4 per thread.
  auto stage = [&](int blk, int buf) {
    const int s0 = blk * 32;
#pragma unroll
    for (int i = 0; i < 4; ++i) {
      const int chunk = tid + i * 128;          // 0..511
      // K tile: 32 key rows x 128 feats
      const int krow = chunk >> 4;
      const int kc0  = (chunk & 15) * 8;
      cp16_g2l(&ks[buf][krow * 128 + kc0],
               Kb + (size_t)(s0 + krow) * D_K + kc0);
      // V^T tile: 128 n rows x 32 keys (padded stride 40)
      const int nrow = chunk >> 2;
      const int vc0  = (chunk & 3) * 8;
      cp16_g2l(&vsT[buf][nrow * 40 + vc0],
               VbT + (size_t)nrow * SEQ + s0 + vc0);
    }
  };

  // Resident Q A-fragments: 16x128 as 4 chunks of K=32, b128 loads.
  AFrag aq[4];
#pragma unroll
  for (int kb = 0; kb < 4; ++kb) {
    const _Float16* qrow = Qb + (size_t)(q0 + ln) * D_K + kb * 32;
    aq[kb].q[0] = *(const uint4*)(qrow + (hi ? 8 : 0));
    aq[kb].q[1] = *(const uint4*)(qrow + (hi ? 24 : 16));
  }

  CFrag o[8];
#pragma unroll
  for (int t = 0; t < 8; ++t)
#pragma unroll
    for (int r = 0; r < 8; ++r) o[t].f[r] = 0.0f;
  float mrow[8], lrow[8];
#pragma unroll
  for (int r = 0; r < 8; ++r) { mrow[r] = -1e30f; lrow[r] = 0.0f; }

  const float scale = 0.08838834764831845f;     // 1/sqrt(128)

  stage(0, 0);                                  // prefetch first key block

  for (int ib = 0; ib < nb; ++ib) {
    wait_async_all();                           // own async loads for cur buf
    __syncthreads();                            // all loads visible, all prev
                                                // reads of next buf finished
    if (ib + 1 < nb) stage(ib + 1, (ib + 1) & 1);

    const int s0k = ib * 32;
    const _Float16* kt  = ks[ib & 1];
    const _Float16* vtT = vsT[ib & 1];

    // ---- scores: S = Q x K^T, two 16-col subtiles, b128 LDS frags ----
    CFrag sc0, sc1;
#pragma unroll
    for (int r = 0; r < 8; ++r) { sc0.f[r] = 0.0f; sc1.f[r] = 0.0f; }
#pragma unroll
    for (int kb = 0; kb < 4; ++kb) {
      AFrag bk0, bk1;
      const int kbase = kb * 32 + (hi ? 8 : 0);
      const _Float16* k0p = kt + ln * 128 + kbase;
      const _Float16* k1p = kt + (16 + ln) * 128 + kbase;
      bk0.q[0] = *(const uint4*)k0p;
      bk0.q[1] = *(const uint4*)(k0p + 16);
      bk1.q[0] = *(const uint4*)k1p;
      bk1.q[1] = *(const uint4*)(k1p + 16);
      sc0.v = __builtin_amdgcn_wmma_f32_16x16x32_f16(false, aq[kb].v, false,
                    bk0.v, (short)0, sc0.v, false, false);
      sc1.v = __builtin_amdgcn_wmma_f32_16x16x32_f16(false, aq[kb].v, false,
                    bk1.v, (short)0, sc1.v, false, false);
    }

    // ---- online softmax (width-16 xor reduce == per-row reduce) ----
#pragma unroll
    for (int r = 0; r < 8; ++r) {
      const int row = q0 + r + (hi ? 8 : 0);
      float s0 = (s0k + ln      <= row) ? sc0.f[r] * scale : -1e30f;
      float s1 = (s0k + 16 + ln <= row) ? sc1.f[r] * scale : -1e30f;
      float mx = fmaxf(s0, s1);
#pragma unroll
      for (int off = 1; off < 16; off <<= 1)
        mx = fmaxf(mx, __shfl_xor(mx, off, 16));
      const float mn    = fmaxf(mrow[r], mx);
      const float alpha = __expf(mrow[r] - mn);
      const float p0 = __expf(s0 - mn);
      const float p1 = __expf(s1 - mn);
      float sum = p0 + p1;
#pragma unroll
      for (int off = 1; off < 16; off <<= 1)
        sum += __shfl_xor(sum, off, 16);
      lrow[r] = lrow[r] * alpha + sum;
      mrow[r] = mn;
#pragma unroll
      for (int t = 0; t < 8; ++t) o[t].f[r] *= alpha;
      const int prow = r + (hi ? 8 : 0);
      ps[wave][prow * 40 + ln]      = (_Float16)p0;
      ps[wave][prow * 40 + 16 + ln] = (_Float16)p1;
    }
    // wave-internal ordering of the P bounce (wave-private LDS region)
    asm volatile("s_wait_dscnt 0" ::: "memory");

    AFrag pa;                                   // P as A-fragment (16x32)
    const _Float16* pp = &ps[wave][ln * 40];
    pa.q[0] = *(const uint4*)(pp + (hi ? 8 : 0));
    pa.q[1] = *(const uint4*)(pp + (hi ? 24 : 16));

    // ---- O += P x V; V frags are contiguous rows of V^T -> b128 reads ----
#pragma unroll
    for (int t = 0; t < 8; ++t) {
      AFrag bv;                                 // B[k][n] = V^T[t*16+n][k]
      const _Float16* vp = vtT + (t * 16 + ln) * 40 + (hi ? 8 : 0);
      bv.q[0] = *(const uint4*)vp;
      bv.q[1] = *(const uint4*)(vp + 16);
      o[t].v = __builtin_amdgcn_wmma_f32_16x16x32_f16(false, pa.v, false,
                    bv.v, (short)0, o[t].v, false, false);
    }
  }

  float* Ob = Out + ((size_t)b * SEQ + q0) * D_K;
#pragma unroll
  for (int r = 0; r < 8; ++r) {
    const float inv = 1.0f / lrow[r];
#pragma unroll
    for (int t = 0; t < 8; ++t)
      Ob[(size_t)(r + (hi ? 8 : 0)) * D_K + t * 16 + ln] = o[t].f[r] * inv;
  }
}

// ---------------------------------------------------------------------------
extern "C" void kernel_launch(void* const* d_in, const int* in_sizes, int n_in,
                              void* d_out, int out_size, void* d_ws,
                              size_t ws_size, hipStream_t stream)
{
  const float* X  = (const float*)d_in[0];
  const float* Wq = (const float*)d_in[1];
  const float* Wk = (const float*)d_in[2];
  const float* Wv = (const float*)d_in[3];
  float* out = (float*)d_out;

  _Float16* Qh = (_Float16*)d_ws;               // 3 x 16384 x 128 f16 = 12.6MB
  _Float16* Kh = Qh + (size_t)MTOT * D_K;
  _Float16* Vt = Kh + (size_t)MTOT * D_K;       // V^T[b][128][2048]

  qkv_project_kernel<<<dim3(MTOT / 16, 3), 256, 0, stream>>>(
      X, Wq, Wk, Wv, Qh, Kh, Vt);
  flash_attn_kernel<<<dim3(SEQ / 16 / 4, BATCH), 128, 0, stream>>>(
      Qh, Kh, Vt, out);
}
